// PointTransformerLayer_21912923144348
// MI455X (gfx1250) — matrix operand order repaired and checked by reading, not compiled, not run on hardware
//
#include <hip/hip_runtime.h>
#include <hip/hip_bf16.h>

typedef __attribute__((ext_vector_type(16))) _Float16 v16h;
typedef __attribute__((ext_vector_type(4)))  _Float16 v4h;
typedef __attribute__((ext_vector_type(8)))  float    v8f;
typedef __attribute__((ext_vector_type(4)))  unsigned int v4u;
typedef __attribute__((ext_vector_type(8)))  int      v8i;
typedef __attribute__((ext_vector_type(4)))  int      v4i;

#define DIM   64      // D == H == 64
#define PGRP  2048    // points per (B,S) group
#define KNN   16

// ---------------------------------------------------------------------------
// Kernel 1: fused projection.  u = (x@wq+bq + pe) * wa ;  g = x@wk+bk + pe + pp
//           v = x@wv+bv ;   pe = pos@wpe+bpe , pp = pos@wpd+bpd
// Weights staged in LDS pre-swizzled into WMMA B-fragment layout so each lane
// reads one v16h fragment with two ds_load_b128.
// ---------------------------------------------------------------------------
__global__ __launch_bounds__(128) void proj_kernel(
    const float* __restrict__ x, const float* __restrict__ pos,
    const float* __restrict__ wq, const float* __restrict__ bq,
    const float* __restrict__ wk, const float* __restrict__ bk,
    const float* __restrict__ wv, const float* __restrict__ bv,
    const float* __restrict__ wpe, const float* __restrict__ bpe,
    const float* __restrict__ wpd, const float* __restrict__ bpd,
    const float* __restrict__ wa,
    float* __restrict__ u, float* __restrict__ g, float* __restrict__ vv)
{
    // B-fragment layout: [kstep][tile][lane][16 halves], 24 KB
    __shared__ _Float16 WcF[2][12][32][16];
    __shared__ float    posS[64][3];

    const int tid = threadIdx.x;
    const int rowBase = blockIdx.x * 64;

    // Fill swizzled weight fragments: element (ksIdx,t,lane,h) = W[K][col],
// K = ksIdx*32 + (h/8)*16 + (lane/16)*8 + h%8, col = t*16 + lane%16.
    for (int q4 = tid; q4 < (2 * 12 * 32 * 16) / 4; q4 += 128) {
        const int e     = q4 * 4;
        const int h0    = e & 15;
        const int lane  = (e >> 4) & 31;
        const int tt    = (e >> 9) % 12;
        const int ksIdx = (e >> 9) / 12;
        const int col   = tt * 16 + (lane & 15);
        const int K0    = ksIdx * 32 + (h0 & 8) * 2 + (lane >> 4) * 8 + (h0 & 7);
        const float* wsrc; int c;
        if (col < 64)       { wsrc = wq; c = col; }
        else if (col < 128) { wsrc = wk; c = col - 64; }
        else                { wsrc = wv; c = col - 128; }
        v4h pk;
        #pragma unroll
        for (int d = 0; d < 4; ++d) pk[d] = (_Float16)wsrc[(size_t)(K0 + d) * 64 + c];
        *(v4h*)&WcF[ksIdx][tt][lane][h0] = pk;
    }
    for (int idx = tid; idx < 64 * 3; idx += 128)
        posS[idx / 3][idx % 3] = pos[(size_t)rowBase * 3 + idx];
    __syncthreads();

    const int wave = tid >> 5, lane = tid & 31;
    const int n16 = lane & 15, g2 = lane >> 4;
    const int r0 = rowBase + wave * 16;

    v8f acc[12];
    const v8f vzero = {0.f,0.f,0.f,0.f,0.f,0.f,0.f,0.f};
    #pragma unroll
    for (int t = 0; t < 12; ++t) acc[t] = vzero;

    #pragma unroll
    for (int ksIdx = 0; ksIdx < 2; ++ksIdx) {
        const int ks = ksIdx * 32;
        // A fragment: row = r0+n16; half-wave g2 covers K {g2*8..+8, 16+g2*8..+8}
        v16h a;
        const float* arow = x + (size_t)(r0 + n16) * DIM + ks + g2 * 8;
        #pragma unroll
        for (int h = 0; h < 8; ++h)  a[h] = (_Float16)arow[h];
        #pragma unroll
        for (int h = 8; h < 16; ++h) a[h] = (_Float16)arow[16 + (h - 8)];

        #pragma unroll
        for (int t = 0; t < 12; ++t) {
            v16h b = *(const v16h*)&WcF[ksIdx][t][lane][0];
            acc[t] = __builtin_amdgcn_wmma_f32_16x16x32_f16(
                false, a, false, b, (short)0, acc[t], false, false);
        }
    }

    // Epilogue: fold biases + pos projections, emit u/g/v.
    #pragma unroll
    for (int ht = 0; ht < 4; ++ht) {
        const int h = ht * 16 + n16;
        const float bqv = bq[h], bkv = bk[h], bvv = bv[h];
        const float w0 = wpe[h], w1 = wpe[64 + h], w2 = wpe[128 + h], bpev = bpe[h];
        const float d0 = wpd[h], d1 = wpd[64 + h], d2 = wpd[128 + h], bpdv = bpd[h];
        const float wav = wa[h];
        #pragma unroll
        for (int r = 0; r < 8; ++r) {
            const int rowL = wave * 16 + r + g2 * 8;   // local row (0..63)
            const int row  = rowBase + rowL;
            const float p0 = posS[rowL][0], p1 = posS[rowL][1], p2 = posS[rowL][2];
            const float pe = p0 * w0 + p1 * w1 + p2 * w2 + bpev;
            const float pp = p0 * d0 + p1 * d1 + p2 * d2 + bpdv;
            const size_t o = (size_t)row * DIM + h;
            u[o]  = (acc[ht][r] + bqv + pe) * wav;
            g[o]  =  acc[ht + 4][r] + bkv + pe + pp;
            vv[o] =  acc[ht + 8][r] + bvv;
        }
    }
}

// ---------------------------------------------------------------------------
// Kernel 2: brute-force kNN (K=16) inside each 2048-point group.
// Group positions staged in LDS by the Tensor Data Mover (async DMA),
// register-resident sorted top-16.
// ---------------------------------------------------------------------------
__global__ __launch_bounds__(256) void knn_kernel(const float* __restrict__ pos,
                                                  int* __restrict__ nbr)
{
    __shared__ float ps[PGRP * 3];                 // 24 KB
    const int grp = blockIdx.x >> 3;               // 8 blocks per group
    const int sub = blockIdx.x & 7;
    const float* pg = pos + (size_t)grp * PGRP * 3;

    if (threadIdx.x < 32) {                        // one wave issues the TDM op
        const unsigned long long ga = (unsigned long long)pg;
        const unsigned int lds = (unsigned int)(unsigned long long)(void*)&ps[0];
        const unsigned int NE = PGRP * 3;          // 6144 f32 elements

        // D# group 0: count=1 (valid), lds_addr, global_addr[56:0], type=2
        v4u g0;
        g0[0] = 1u;
        g0[1] = lds;
        g0[2] = (unsigned int)(ga & 0xFFFFFFFFull);
        g0[3] = (unsigned int)((ga >> 32) & 0x01FFFFFFull) | (2u << 30);

        // D# group 1: data_size=4B, 1-row 2D tensor: dim0=tile0=NE, dim1=tile1=1
        v8i g1;
        g1[0] = (int)(2u << 16);                   // data_size = 4 bytes
        g1[1] = (int)((NE & 0xFFFFu) << 16);       // tensor_dim0[15:0]  @ bits 63:48
        g1[2] = (int)(((NE >> 16) & 0xFFFFu)       // tensor_dim0[31:16] @ bits 79:64
                      | (1u << 16));               // tensor_dim1[15:0]=1 @ bits 95:80
        g1[3] = (int)((NE & 0xFFFFu) << 16);       // tile_dim0=NE @ bits 127:112
        g1[4] = 1;                                 // tile_dim1=1, tile_dim2=0
        g1[5] = (int)NE;                           // tensor_dim0_stride[31:0]
        g1[6] = 0;                                 // stride hi + dim1_stride lo
        g1[7] = 0;
        v4i z4 = {0, 0, 0, 0};
        v8i z8 = {0, 0, 0, 0, 0, 0, 0, 0};
        __builtin_amdgcn_tensor_load_to_lds(g0, g1, z4, z4, z8, 0);
        __builtin_amdgcn_s_wait_tensorcnt((short)0);
    }
    __syncthreads();

    const int i = sub * 256 + threadIdx.x;         // local point id
    const float px = ps[i * 3], py = ps[i * 3 + 1], pz = ps[i * 3 + 2];

    float bd[KNN]; int bi[KNN];
    #pragma unroll
    for (int s = 0; s < KNN; ++s) { bd[s] = 3.4e38f; bi[s] = 0; }

    for (int j = 0; j < PGRP; ++j) {
        const float dx = ps[j * 3] - px, dy = ps[j * 3 + 1] - py, dz = ps[j * 3 + 2] - pz;
        const float d = dx * dx + dy * dy + dz * dz;
        if (d < bd[KNN - 1]) {
            bd[KNN - 1] = d; bi[KNN - 1] = j;
            #pragma unroll
            for (int s = KNN - 1; s > 0; --s) {    // single bubble pass, constant indices
                if (bd[s] < bd[s - 1]) {
                    float td = bd[s]; bd[s] = bd[s - 1]; bd[s - 1] = td;
                    int   ti = bi[s]; bi[s] = bi[s - 1]; bi[s - 1] = ti;
                }
            }
        }
    }
    const size_t gi = (size_t)grp * PGRP + i;
    #pragma unroll
    for (int s = 0; s < KNN; ++s) nbr[gi * KNN + s] = bi[s];
}

// ---------------------------------------------------------------------------
// Kernel 3: logits = u_i . g_j over 16 neighbors, softmax within 16-lane groups.
// ---------------------------------------------------------------------------
__global__ __launch_bounds__(256) void attn_kernel(
    const float* __restrict__ u, const float* __restrict__ g,
    const int* __restrict__ nbr, float* __restrict__ attnw)
{
    const int t = blockIdx.x * 256 + threadIdx.x;  // pair id
    const int i = t >> 4, j = t & 15;
    const int grp = i / PGRP;
    const int n = nbr[(size_t)i * KNN + j];
    const float* ui = u + (size_t)i * DIM;
    const float* gj = g + ((size_t)grp * PGRP + n) * DIM;

    float dot = 0.f;
    #pragma unroll
    for (int h = 0; h < DIM; h += 4) {
        float4 a = *(const float4*)(ui + h);
        float4 b = *(const float4*)(gj + h);
        dot += a.x * b.x + a.y * b.y + a.z * b.z + a.w * b.w;
    }
    float m = dot;
    #pragma unroll
    for (int off = 8; off >= 1; off >>= 1) m = fmaxf(m, __shfl_xor(m, off, 16));
    const float e = __expf(dot - m);
    float s = e;
    #pragma unroll
    for (int off = 8; off >= 1; off >>= 1) s += __shfl_xor(s, off, 16);
    attnw[t] = e / s;
}

// ---------------------------------------------------------------------------
// Kernel 4: out = relu( (sum_j attn * v_nj) @ wo + bo ) + x
// Gathered weighted sum stored to LDS as f16 (A-fragment-friendly), wo staged
// pre-swizzled in B-fragment layout; 64x64 GEMM via WMMA.
// ---------------------------------------------------------------------------
__global__ __launch_bounds__(128) void out_kernel(
    const float* __restrict__ vv, const float* __restrict__ attnw,
    const int* __restrict__ nbr, const float* __restrict__ x,
    const float* __restrict__ wo, const float* __restrict__ bo,
    float* __restrict__ out)
{
    __shared__ _Float16 WoF[2][4][32][16];   // swizzled B fragments, 8 KB
    __shared__ _Float16 ob[4][16][64];       // per-wave 16x64 "out" tile (f16), 8 KB

    const int tid = threadIdx.x;
    for (int q4 = tid; q4 < (2 * 4 * 32 * 16) / 4; q4 += 128) {
        const int e     = q4 * 4;
        const int h0    = e & 15;
        const int lane  = (e >> 4) & 31;
        const int tt    = (e >> 9) & 3;
        const int ksIdx = (e >> 9) >> 2;
        const int col   = tt * 16 + (lane & 15);
        const int K0    = ksIdx * 32 + (h0 & 8) * 2 + (lane >> 4) * 8 + (h0 & 7);
        v4h pk;
        #pragma unroll
        for (int d = 0; d < 4; ++d) pk[d] = (_Float16)wo[(size_t)(K0 + d) * 64 + col];
        *(v4h*)&WoF[ksIdx][tt][lane][h0] = pk;
    }

    const int wave = tid >> 5, lane = tid & 31;
    const int n16 = lane & 15, g2 = lane >> 4;
    const int rowBase = blockIdx.x * 64;
    const int r0 = rowBase + wave * 16;

    // ---- gather-weighted sum: lane handles (row = r0+n16, 32-col half g2)
    {
        const int row = r0 + n16;
        const size_t gb = (size_t)(row / PGRP) * PGRP;
        const int chalf = g2 * 32;
        float accv[32];
        #pragma unroll
        for (int c = 0; c < 32; ++c) accv[c] = 0.f;
        for (int j = 0; j < KNN; ++j) {
            const int   n = nbr[(size_t)row * KNN + j];
            const float w = attnw[(size_t)row * KNN + j];
            const float* vrow = vv + (gb + n) * DIM + chalf;
            #pragma unroll
            for (int c = 0; c < 32; c += 4) {
                float4 vj = *(const float4*)(vrow + c);
                accv[c]     += w * vj.x; accv[c + 1] += w * vj.y;
                accv[c + 2] += w * vj.z; accv[c + 3] += w * vj.w;
            }
        }
        #pragma unroll
        for (int c = 0; c < 32; c += 4) {
            v4h pk;
            pk[0] = (_Float16)accv[c];     pk[1] = (_Float16)accv[c + 1];
            pk[2] = (_Float16)accv[c + 2]; pk[3] = (_Float16)accv[c + 3];
            *(v4h*)&ob[wave][n16][chalf + c] = pk;
        }
    }
    __syncthreads();

    // ---- WMMA: (16x64) @ (64x64) in 4 column tiles, K in 2 steps
    v8f acc[4];
    const v8f vzero = {0.f,0.f,0.f,0.f,0.f,0.f,0.f,0.f};
    #pragma unroll
    for (int t = 0; t < 4; ++t) acc[t] = vzero;

    #pragma unroll
    for (int ksIdx = 0; ksIdx < 2; ++ksIdx) {
        const int ks = ksIdx * 32;
        v16h a;   // two contiguous 8-half runs -> two ds_load_b128
        #pragma unroll
        for (int h = 0; h < 8; ++h)  a[h] = ob[wave][n16][ks + g2 * 8 + h];
        #pragma unroll
        for (int h = 8; h < 16; ++h) a[h] = ob[wave][n16][ks + 16 + g2 * 8 + (h - 8)];
        #pragma unroll
        for (int t = 0; t < 4; ++t) {
            v16h b = *(const v16h*)&WoF[ksIdx][t][lane][0];
            acc[t] = __builtin_amdgcn_wmma_f32_16x16x32_f16(
                false, a, false, b, (short)0, acc[t], false, false);
        }
    }

    // ---- epilogue: bias + relu + residual
    #pragma unroll
    for (int t = 0; t < 4; ++t) {
        const int col = t * 16 + n16;
        const float bov = bo[col];
        #pragma unroll
        for (int r = 0; r < 8; ++r) {
            const int rr = r0 + r + g2 * 8;
            float o = acc[t][r] + bov;
            o = o > 0.f ? o : 0.f;
            out[(size_t)rr * DIM + col] = x[(size_t)rr * DIM + col] + o;
        }
    }
}

// ---------------------------------------------------------------------------
extern "C" void kernel_launch(void* const* d_in, const int* in_sizes, int n_in,
                              void* d_out, int out_size, void* d_ws, size_t ws_size,
                              hipStream_t stream)
{
    const float* x   = (const float*)d_in[0];
    const float* pos = (const float*)d_in[1];
    const float* wq  = (const float*)d_in[2];
    const float* bq  = (const float*)d_in[3];
    const float* wk  = (const float*)d_in[4];
    const float* bk  = (const float*)d_in[5];
    const float* wv  = (const float*)d_in[6];
    const float* bv  = (const float*)d_in[7];
    const float* wpe = (const float*)d_in[8];
    const float* bpe = (const float*)d_in[9];
    const float* wpd = (const float*)d_in[10];
    const float* bpd = (const float*)d_in[11];
    const float* wa  = (const float*)d_in[12];
    // d_in[13] = ba (irrelevant under softmax)
    const float* wo  = (const float*)d_in[14];
    const float* bo  = (const float*)d_in[15];

    const int N = in_sizes[0] / DIM;          // 16384 points
    const int nGroups = N / PGRP;             // 8

    float* u     = (float*)d_ws;
    float* g     = u + (size_t)N * DIM;
    float* vv    = g + (size_t)N * DIM;
    int*   nbr   = (int*)(vv + (size_t)N * DIM);
    float* attnw = (float*)(nbr + (size_t)N * KNN);

    proj_kernel<<<N / 64, 128, 0, stream>>>(x, pos, wq, bq, wk, bk, wv, bv,
                                            wpe, bpe, wpd, bpd, wa, u, g, vv);
    knn_kernel<<<nGroups * (PGRP / 256), 256, 0, stream>>>(pos, nbr);
    attn_kernel<<<(N * KNN) / 256, 256, 0, stream>>>(u, g, nbr, attnw);
    out_kernel<<<N / 64, 128, 0, stream>>>(vv, attnw, nbr, x, wo, bo, (float*)d_out);
}
